// MoshiFlexibleLinearFL_49804440765154
// MI455X (gfx1250) — compile-verified, matrix operand order checked by reading, not run on hardware
//
#include <hip/hip_runtime.h>

typedef __attribute__((ext_vector_type(16))) __bf16 v16bf;
typedef __attribute__((ext_vector_type(8)))  __bf16 v8bf;
typedef __attribute__((ext_vector_type(4)))  __bf16 v4bf;
typedef __attribute__((ext_vector_type(8)))  float  v8f;

#define NUM_LAYERS 16
#define IN_SIZE    1024
#define OUT_SIZE   1024
#define BATCH      64
#define SEQ        128

#define BK    32          // K-block staged per iteration (one bf16 WMMA K)
#define LSTR  56          // LDS row stride in bf16 (112B: 16B-aligned, bank-conflict-free)
#define NBLK  128         // output columns per block (8 waves x 16)

// Assemble a v16bf WMMA fragment from the LDS tile.
// Per ISA 16-bit A-matrix 16x32 layout: a lane holds K = [base, base+8) and
// [base+16, base+24) with base = (lane>=16 ? 8 : 0). Two ds_load_b128's.
__device__ __forceinline__ v16bf load_frag(const __bf16* p) {
    v8bf c0 = *(const v8bf*)(p);        // K = base .. base+7
    v8bf c1 = *(const v8bf*)(p + 16);   // K = base+16 .. base+23
    return __builtin_shufflevector(c0, c1, 0, 1, 2, 3, 4, 5, 6, 7,
                                           8, 9, 10, 11, 12, 13, 14, 15);
}

// Split a float4 into bf16 hi + bf16 residual (lo), returned by value.
struct Split4 { v4bf h; v4bf l; };
__device__ __forceinline__ Split4 split4(float4 v) {
    Split4 r;
    float f0 = v.x, f1 = v.y, f2 = v.z, f3 = v.w;
    __bf16 h0 = (__bf16)f0, h1 = (__bf16)f1, h2 = (__bf16)f2, h3 = (__bf16)f3;
    r.h[0] = h0; r.h[1] = h1; r.h[2] = h2; r.h[3] = h3;
    r.l[0] = (__bf16)(f0 - (float)h0);
    r.l[1] = (__bf16)(f1 - (float)h1);
    r.l[2] = (__bf16)(f2 - (float)h2);
    r.l[3] = (__bf16)(f3 - (float)h3);
    return r;
}

__global__ __launch_bounds__(256)
void flexlin_bf16x3_kernel(const float* __restrict__ x,
                           const int*   __restrict__ layer_idx,
                           const float* __restrict__ w,
                           float*       __restrict__ out) {
    __shared__ __bf16 Xhi[BATCH * LSTR];   // 64 x 32 (stride 56)
    __shared__ __bf16 Xlo[BATCH * LSTR];
    __shared__ __bf16 Whi[NBLK * LSTR];    // 128 x 32 (stride 56)
    __shared__ __bf16 Wlo[NBLK * LSTR];

    const int s  = blockIdx.x;                 // sequence position
    const int n0 = blockIdx.y * NBLK;          // first output column of block
    const int l  = layer_idx[s];               // uniform -> scalar load

    const int tid   = threadIdx.x;
    const int wave  = tid >> 5;                // 0..7
    const int lane  = tid & 31;
    const int lrow  = lane & 15;               // M (A) / N (B) index
    const int khalf = lane >> 4;               // 0 or 1
    const int base  = khalf * 8;               // fragment K base

    const float* xs = x + (size_t)s * IN_SIZE; // row b at + b*SEQ*IN_SIZE
    const float* wg = w + ((size_t)l * OUT_SIZE + n0) * IN_SIZE;

    v8f acc[4] = {v8f{}, v8f{}, v8f{}, v8f{}};

    for (int k0 = 0; k0 < IN_SIZE; k0 += BK) {
        // ---- stage X tile: 64 rows x 32 cols, convert fp32 -> (hi, lo) bf16
#pragma unroll
        for (int i = 0; i < 2; ++i) {
            int idx = tid + i * 256;
            int row = idx >> 3;
            int col = (idx & 7) << 2;
            float4 v = *(const float4*)(xs + (size_t)row * (SEQ * IN_SIZE) + k0 + col);
            Split4 sp = split4(v);
            *(v4bf*)&Xhi[row * LSTR + col] = sp.h;
            *(v4bf*)&Xlo[row * LSTR + col] = sp.l;
        }
        // ---- stage W tile: 128 rows x 32 cols
#pragma unroll
        for (int i = 0; i < 4; ++i) {
            int idx = tid + i * 256;
            int row = idx >> 3;
            int col = (idx & 7) << 2;
            const float* gp = wg + (size_t)row * IN_SIZE + k0 + col;
            float4 v = *(const float4*)gp;
            if (k0 + BK < IN_SIZE)
                __builtin_prefetch(gp + BK, 0, 0);   // global_prefetch_b8 next K-block
            Split4 sp = split4(v);
            *(v4bf*)&Whi[row * LSTR + col] = sp.h;
            *(v4bf*)&Wlo[row * LSTR + col] = sp.l;
        }
        __syncthreads();

        // ---- compute: per wave, B frags for its 16 columns, 4 M-tiles
        const int brow = wave * 16 + lrow;
        v16bf bhi = load_frag(&Whi[brow * LSTR + base]);
        v16bf blo = load_frag(&Wlo[brow * LSTR + base]);
#pragma unroll
        for (int m = 0; m < 4; ++m) {
            const int arow = m * 16 + lrow;
            v16bf ahi = load_frag(&Xhi[arow * LSTR + base]);
            v16bf alo = load_frag(&Xlo[arow * LSTR + base]);
            // 3-pass fp32 emulation: hi*hi + hi*lo + lo*hi (lo*lo ~2^-16, dropped)
            acc[m] = __builtin_amdgcn_wmma_f32_16x16x32_bf16(
                false, ahi, false, bhi, (short)0, acc[m], false, false);
            acc[m] = __builtin_amdgcn_wmma_f32_16x16x32_bf16(
                false, ahi, false, blo, (short)0, acc[m], false, false);
            acc[m] = __builtin_amdgcn_wmma_f32_16x16x32_bf16(
                false, alo, false, bhi, (short)0, acc[m], false, false);
        }
        __syncthreads();
    }

    // ---- epilogue: C/D layout — lane<16: M=r, lane>=16: M=8+r; N = lane&15
    const int o = n0 + wave * 16 + lrow;
#pragma unroll
    for (int m = 0; m < 4; ++m) {
        const int bb = m * 16 + khalf * 8;
#pragma unroll
        for (int r = 0; r < 8; ++r) {
            out[((size_t)(bb + r) * SEQ + s) * OUT_SIZE + o] = acc[m][r];
        }
    }
}

extern "C" void kernel_launch(void* const* d_in, const int* in_sizes, int n_in,
                              void* d_out, int out_size, void* d_ws, size_t ws_size,
                              hipStream_t stream) {
    (void)in_sizes; (void)n_in; (void)out_size; (void)d_ws; (void)ws_size;
    const float* x   = (const float*)d_in[0];
    const int*   idx = (const int*)d_in[1];
    const float* w   = (const float*)d_in[2];
    float*       out = (float*)d_out;

    dim3 grid(SEQ, OUT_SIZE / NBLK);   // (128, 8)
    dim3 block(256);                   // 8 waves (wave32)
    flexlin_bf16x3_kernel<<<grid, block, 0, stream>>>(x, idx, w, out);
}
